// BinaryEdgeReconstructionLayer_20306605376128
// MI455X (gfx1250) — matrix operand order repaired and checked by reading, not compiled, not run on hardware
//
#include <hip/hip_runtime.h>
#include <hip/hip_bf16.h>
#include <math.h>

// ---------------------------------------------------------------------------
// Problem constants (from reference): B=4, N=48, HID=256, COND=32, H=8, HD=32
// ---------------------------------------------------------------------------
#define Bsz   4
#define Nn    48
#define NN    (Nn * Nn)          // 2304
#define NNP   (NN + 64)          // padded kl extent for repacked V
#define HID   256
#define CND   32
#define NH    8
#define HD    32
#define Mrows (Bsz * NN)         // 9216 edge rows
#define Mnode (Bsz * Nn)         // 192 node rows
#define FFN   1024
#define SCALE 0.17677669529663687f  // 1/sqrt(32)

typedef __attribute__((ext_vector_type(16))) __bf16 v16bf;
typedef __attribute__((ext_vector_type(8)))  float  v8f;
typedef __attribute__((ext_vector_type(4)))  int    v4i;

#ifndef __has_builtin
#define __has_builtin(x) 0
#endif
#if defined(__AMDGCN__) && __has_builtin(__builtin_amdgcn_global_load_async_to_lds_b128)
#define HAVE_ASYNC_LDS 1
#else
#define HAVE_ASYNC_LDS 0
#endif

// d_in index map. Assumes setup_inputs() dict order, params in insertion
// order, each _lin flattened as {w, b}.
enum {
  IN_EF = 0, IN_NF, IN_EM, IN_NM, IN_COND,
  P_Q_W, P_Q_B, P_K_W, P_K_B, P_V_W, P_V_B,
  P_CP_W, P_CP_B, P_CG_W, P_CG_B,
  P_TB_W, P_TB_B,          // tri_bias: provably a no-op under softmax -> skipped
  P_TG_W, P_TG_B, P_TO_W, P_TO_B,
  P_NP_W, P_NP_B, P_N2K_W, P_N2K_B, P_N2V_W, P_N2V_B,
  P_E2Q_W, P_E2Q_B, P_N2EO_W, P_N2EO_B,
  P_LN1G, P_LN1B, P_LN2G, P_LN2B, P_LN3G, P_LN3B,
  P_FFN1_W, P_FFN1_B, P_FFN2_W, P_FFN2_B
};

// ---------------------------------------------------------------------------
// WMMA fragment index helpers (CDNA5 ISA 7.12.2, 16-bit layouts, wave32)
// A 16x32: lane L -> row M=L&15; elem j -> K = half*16+j (this code's contiguous order, not the ISA table order)
// B 32x16: lane L -> col N=L&15; elem j -> K = half*16+j (same contiguous order as A, not the ISA table order)
// C/D f32: lane L -> col N=L&15; VGPR r -> row M = r + half*8
// ---------------------------------------------------------------------------
__device__ __forceinline__ int a_kk(int j, int half) {
  return half * 16 + j;
}

// ---------------------------------------------------------------------------
// bf16 WMMA GEMM: D[M,N] = act(A[M,K] @ W + bias), W stored TRANSPOSED [N,K].
// Workgroup = 4 waves = 64x64 output tile; each wave does 16 rows x 64 cols.
// A tile (64x32) is staged in LDS once per K-step (async copy when available)
// and shared by all 4 waves. ACT: 0=none, 1=sigmoid, 2=exact GELU.
// ---------------------------------------------------------------------------
template <int ACT>
__global__ __launch_bounds__(128) void gemm_bf16_kernel(
    const __bf16* __restrict__ A, const __bf16* __restrict__ Wt,
    const float* __restrict__ bias, float* __restrict__ Cf,
    __bf16* __restrict__ Cb, int M, int N, int K)
{
  __shared__ __align__(16) __bf16 sA[64 * 32];
  const int tid = threadIdx.x;
  const int lane = tid & 31, wave = tid >> 5;
  const int m = lane & 15, half = lane >> 4;
  const int row0 = blockIdx.x * 64;
  const int col0 = blockIdx.y * 64;

  v8f acc[4] = {v8f{}, v8f{}, v8f{}, v8f{}};

  for (int k0 = 0; k0 < K; k0 += 32) {
    // --- stage A tile 64x32 (4 KB) into LDS: 256 16-byte chunks / 128 thr ---
#pragma unroll
    for (int c = 0; c < 2; ++c) {
      const int chunk = tid + c * 128;          // 0..255
      const int r = chunk >> 2, q = chunk & 3;  // row, 8-elem group
      const __bf16* g = A + (size_t)(row0 + r) * K + k0 + q * 8;
      __bf16* l = sA + r * 32 + q * 8;
#if HAVE_ASYNC_LDS
      __builtin_amdgcn_global_load_async_to_lds_b128((v4i*)g, (v4i*)l, 0, 0);
#else
      *(uint4*)l = *(const uint4*)g;
#endif
    }
#if HAVE_ASYNC_LDS
#if __has_builtin(__builtin_amdgcn_s_wait_asynccnt)
    __builtin_amdgcn_s_wait_asynccnt(0);
#else
    asm volatile("s_wait_asynccnt 0" ::: "memory");
#endif
#endif
    __syncthreads();

    v16bf af;
#pragma unroll
    for (int j = 0; j < 16; ++j)
      af[j] = sA[(wave * 16 + m) * 32 + a_kk(j, half)];  // 2x ds_load_b128

#pragma unroll
    for (int t4 = 0; t4 < 4; ++t4) {
      const int n = col0 + t4 * 16 + m;
      const __bf16* bp = Wt + (size_t)n * K + k0 + half * 16;
      __builtin_prefetch(bp + 32, 0, 1);  // global_prefetch_b8 (next K-step)
      v16bf bfr;
#pragma unroll
      for (int j = 0; j < 16; ++j) bfr[j] = bp[j];  // 2x global_load_b128
      acc[t4] = __builtin_amdgcn_wmma_f32_16x16x32_bf16(false, af, false, bfr,
                                                        (short)0, acc[t4],
                                                        false, false);
    }
    __syncthreads();
  }

#pragma unroll
  for (int t4 = 0; t4 < 4; ++t4) {
    const int col = col0 + t4 * 16 + m;
    const float bz = bias ? bias[col] : 0.0f;
#pragma unroll
    for (int r = 0; r < 8; ++r) {
      const size_t row = (size_t)(row0 + wave * 16 + half * 8 + r);
      float v = acc[t4][r] + bz;
      if (ACT == 1) v = 1.0f / (1.0f + __expf(-v));
      if (ACT == 2) v = 0.5f * v * (1.0f + erff(v * 0.70710678118654752f));
      if (Cf) Cf[row * N + col] = v;
      if (Cb) Cb[row * N + col] = (__bf16)v;
    }
  }
}

// ---------------------------------------------------------------------------
// Streaming triangle attention. One wave per (b, h, 16-query tile).
// Per k-group: S[16x48] = Q_tile @ K^T (3 WMMAs), softmax over l in LDS,
// out += P @ V (4 WMMAs; V repacked [b,h,d,kl] padded so B-frags are
// contiguous 32B reads, zero-prob padding covers kl >= 48).
// tri_bias omitted: softmax_l(s + c_q) == softmax_l(s) exactly.
// ---------------------------------------------------------------------------
__global__ __launch_bounds__(32) void tri_attn_kernel(
    const __bf16* __restrict__ Q, const __bf16* __restrict__ Km,
    const __bf16* __restrict__ Vt, const float* __restrict__ em,
    float* __restrict__ att_f, __bf16* __restrict__ att_b)
{
  const int lane = threadIdx.x;
  const int m = lane & 15, half = lane >> 4;
  int bid = blockIdx.x;
  const int qt = bid % (NN / 16); bid /= (NN / 16);
  const int h = bid % NH; const int b = bid / NH;

  __shared__ float  sS[16][Nn];
  __shared__ __align__(16) __bf16 sP[16][64];

  v16bf qa;
  {
    const __bf16* qrow = Q + ((size_t)(b * NN + qt * 16 + m)) * HID + h * HD;
#pragma unroll
    for (int j = 0; j < 16; ++j) qa[j] = qrow[a_kk(j, half)];
  }

  v8f acc0 = {}, acc1 = {};

  for (int k = 0; k < Nn; ++k) {
    // --- scores ---
#pragma unroll
    for (int lt = 0; lt < 3; ++lt) {
      const int l = lt * 16 + m;
      v16bf kb;
      const __bf16* kp =
          Km + ((size_t)(b * NN + k * Nn + l)) * HID + h * HD + half * 16;
#pragma unroll
      for (int j = 0; j < 16; ++j) kb[j] = kp[j];
      v8f s = {};
      s = __builtin_amdgcn_wmma_f32_16x16x32_bf16(false, qa, false, kb,
                                                  (short)0, s, false, false);
      const float mb = (em[(size_t)b * NN + k * Nn + l] == 0.0f) ? -1e30f : 0.0f;
#pragma unroll
      for (int r = 0; r < 8; ++r) sS[half * 8 + r][l] = s[r] * SCALE + mb;
    }
    __syncthreads();
    // --- per-row softmax over 48 l values ---
    if (lane < 16) {
      float mx = -1e30f;
      for (int l = 0; l < Nn; ++l) mx = fmaxf(mx, sS[lane][l]);
      float sum = 0.0f;
      for (int l = 0; l < Nn; ++l) {
        float e = __expf(sS[lane][l] - mx);
        sS[lane][l] = e; sum += e;
      }
      const float inv = 1.0f / sum;
      for (int l = 0; l < Nn; ++l) sP[lane][l] = (__bf16)(sS[lane][l] * inv);
#pragma unroll
      for (int l = Nn; l < 64; ++l) sP[lane][l] = (__bf16)0.0f;
    }
    __syncthreads();
    // --- P @ V ---
#pragma unroll
    for (int kc = 0; kc < 2; ++kc) {
      v16bf pa;
#pragma unroll
      for (int j = 0; j < 16; ++j) pa[j] = sP[m][kc * 32 + a_kk(j, half)];
#pragma unroll
      for (int dt = 0; dt < 2; ++dt) {
        const __bf16* vp = Vt +
            ((size_t)((b * NH + h) * HD + dt * 16 + m)) * NNP +
            k * Nn + kc * 32 + half * 16;
        v16bf vb;
#pragma unroll
        for (int j = 0; j < 16; ++j) vb[j] = vp[j];  // contiguous 32B
        if (dt == 0)
          acc0 = __builtin_amdgcn_wmma_f32_16x16x32_bf16(false, pa, false, vb,
                                                         (short)0, acc0, false, false);
        else
          acc1 = __builtin_amdgcn_wmma_f32_16x16x32_bf16(false, pa, false, vb,
                                                         (short)0, acc1, false, false);
      }
    }
  }
#pragma unroll
  for (int r = 0; r < 8; ++r) {
    const size_t row = (size_t)(b * NN + qt * 16 + half * 8 + r);
    const size_t c0 = row * HID + h * HD + m;
    att_f[c0] = acc0[r];      att_f[c0 + 16] = acc1[r];
    att_b[c0] = (__bf16)acc0[r]; att_b[c0 + 16] = (__bf16)acc1[r];
  }
}

// ---------------------------------------------------------------------------
// Small helper kernels
// ---------------------------------------------------------------------------
// Transposing f32 -> bf16 weight conversion: dst[n*K + k] = src[k*N + n]
__global__ void convT_kernel(const float* __restrict__ s,
                             __bf16* __restrict__ d, int K, int N) {
  const int n_tot = K * N;
  for (int t = blockIdx.x * blockDim.x + threadIdx.x; t < n_tot;
       t += gridDim.x * blockDim.x) {
    const int n = t / K, k = t % K;
    d[t] = (__bf16)s[(size_t)k * N + n];
  }
}

// Repack V [row, h*32+d] -> Vt [b][h][d][kl] with kl padded to NNP (zeros).
__global__ void repack_vt_kernel(const __bf16* __restrict__ Vb,
                                 __bf16* __restrict__ Vt) {
  const size_t n_tot = (size_t)Bsz * NH * HD * NNP;
  for (size_t t = (size_t)blockIdx.x * blockDim.x + threadIdx.x; t < n_tot;
       t += (size_t)gridDim.x * blockDim.x) {
    const int kl = (int)(t % NNP);
    size_t rem = t / NNP;
    const int d = (int)(rem % HD); rem /= HD;
    const int h = (int)(rem % NH); const int b = (int)(rem / NH);
    Vt[t] = (kl < NN) ? Vb[((size_t)(b * NN + kl)) * HID + h * HD + d]
                      : (__bf16)0.0f;
  }
}

__global__ __launch_bounds__(256) void cond_prep_kernel(
    const float* __restrict__ cond, const float* __restrict__ wcp,
    const float* __restrict__ bcp, const float* __restrict__ wcg,
    const float* __restrict__ bcg, float* __restrict__ cp,
    float* __restrict__ cg) {
  const int b = blockIdx.x, o = threadIdx.x;
  float s1 = bcp[o], s2 = bcg[o];
  for (int c = 0; c < CND; ++c) {
    const float cv = cond[b * CND + c];
    s1 += cv * wcp[c * HID + o];
    s2 += cv * wcg[c * HID + o];
  }
  cp[b * HID + o] = s1;
  cg[b * HID + o] = 1.0f / (1.0f + __expf(-s2));
}

__global__ void build_cf_kernel(const float* __restrict__ ef,
                                const float* __restrict__ cp,
                                const float* __restrict__ cg,
                                __bf16* __restrict__ cf) {
  const size_t n = (size_t)Mrows * HID;
  for (size_t t = (size_t)blockIdx.x * blockDim.x + threadIdx.x; t < n;
       t += (size_t)gridDim.x * blockDim.x) {
    const int c = (int)(t & (HID - 1));
    const int b = (int)(t >> 8) / NN;
    cf[t] = (__bf16)(ef[t] * cg[b * HID + c] + cp[b * HID + c]);
  }
}

__global__ void mul_to_bf16_kernel(const float* __restrict__ a,
                                   const float* __restrict__ g,
                                   __bf16* __restrict__ d, int n) {
  for (int t = blockIdx.x * blockDim.x + threadIdx.x; t < n;
       t += gridDim.x * blockDim.x)
    d[t] = (__bf16)(a[t] * g[t]);
}

__global__ void build_cn_kernel(const float* __restrict__ nf,
                                const float* __restrict__ cond,
                                __bf16* __restrict__ cn) {
  const int n = Mnode * (HID + CND);
  for (int t = blockIdx.x * blockDim.x + threadIdx.x; t < n;
       t += gridDim.x * blockDim.x) {
    const int row = t / (HID + CND), c = t % (HID + CND);
    const int b = row / Nn;
    const float v = (c < HID) ? nf[row * HID + c] : cond[b * CND + (c - HID)];
    cn[t] = (__bf16)v;
  }
}

// Per-(b,i): si[j] = q[b,i,j]·keys[b,i], sj[j] = q[b,i,j]·keys[b,j]; softmax over j.
__global__ __launch_bounds__(64) void n2e_scores_kernel(
    const float* __restrict__ qe, const float* __restrict__ keys,
    const float* __restrict__ nm, float* __restrict__ ai,
    float* __restrict__ aj) {
  const int b = blockIdx.x / Nn, i = blockIdx.x % Nn;
  const int j = threadIdx.x;
  __shared__ float red[64];
  float si = -1e30f, sj = -1e30f;
  if (j < Nn) {
    const float* qv = qe + ((size_t)((b * Nn + i) * Nn + j)) * HID;
    const float* ki = keys + (size_t)(b * Nn + i) * HID;
    const float* kj = keys + (size_t)(b * Nn + j) * HID;
    float di = 0.0f, dj = 0.0f;
    for (int d = 0; d < HID; ++d) { const float q = qv[d]; di += q * ki[d]; dj += q * kj[d]; }
    si = di * SCALE; sj = dj * SCALE;
    if (nm[b * Nn + j] == 0.0f) si = -1e30f;
    if (nm[b * Nn + i] == 0.0f) sj = -1e30f;
  }
  red[threadIdx.x] = si; __syncthreads();
  for (int s = 32; s > 0; s >>= 1) { if (threadIdx.x < s) red[threadIdx.x] = fmaxf(red[threadIdx.x], red[threadIdx.x + s]); __syncthreads(); }
  const float mi = red[0]; __syncthreads();
  const float ei = (j < Nn) ? __expf(si - mi) : 0.0f;
  red[threadIdx.x] = ei; __syncthreads();
  for (int s = 32; s > 0; s >>= 1) { if (threadIdx.x < s) red[threadIdx.x] += red[threadIdx.x + s]; __syncthreads(); }
  const float sumi = red[0]; __syncthreads();
  red[threadIdx.x] = sj; __syncthreads();
  for (int s = 32; s > 0; s >>= 1) { if (threadIdx.x < s) red[threadIdx.x] = fmaxf(red[threadIdx.x], red[threadIdx.x + s]); __syncthreads(); }
  const float mj = red[0]; __syncthreads();
  const float ej = (j < Nn) ? __expf(sj - mj) : 0.0f;
  red[threadIdx.x] = ej; __syncthreads();
  for (int s = 32; s > 0; s >>= 1) { if (threadIdx.x < s) red[threadIdx.x] += red[threadIdx.x + s]; __syncthreads(); }
  const float sumj = red[0];
  if (j < Nn) {
    ai[blockIdx.x * Nn + j] = ei / sumi;
    aj[blockIdx.x * Nn + j] = ej / sumj;
  }
}

__global__ void build_pre_kernel(const float* __restrict__ ai,
                                 const float* __restrict__ aj,
                                 const float* __restrict__ vals,
                                 const float* __restrict__ em,
                                 __bf16* __restrict__ pre) {
  const size_t n = (size_t)Mrows * HID;
  for (size_t t = (size_t)blockIdx.x * blockDim.x + threadIdx.x; t < n;
       t += (size_t)gridDim.x * blockDim.x) {
    const int d = (int)(t & (HID - 1));
    const int row = (int)(t >> 8);
    const int b = row / NN, rem = row % NN, i = rem / Nn, j = rem % Nn;
    const float a_i = ai[(b * Nn + i) * Nn + j];
    const float a_j = aj[(b * Nn + i) * Nn + j];
    const float v = (a_i * vals[(size_t)(b * Nn + i) * HID + d] +
                     a_j * vals[(size_t)(b * Nn + j) * HID + d]) * em[row];
    pre[t] = (__bf16)v;
  }
}

// Fused residual add + LayerNorm over last dim (256). One 256-thread block/row.
__global__ __launch_bounds__(256) void add_ln_kernel(
    const float* __restrict__ x, const float* __restrict__ res,
    const float* __restrict__ g, const float* __restrict__ bt,
    float* __restrict__ of, __bf16* __restrict__ ob) {
  const int row = blockIdx.x, t = threadIdx.x;
  __shared__ float s1[256];
  const float v = x[(size_t)row * HID + t] + res[(size_t)row * HID + t];
  s1[t] = v; __syncthreads();
  for (int s = 128; s > 0; s >>= 1) { if (t < s) s1[t] += s1[t + s]; __syncthreads(); }
  const float mean = s1[0] * (1.0f / HID); __syncthreads();
  const float dv = v - mean;
  s1[t] = dv * dv; __syncthreads();
  for (int s = 128; s > 0; s >>= 1) { if (t < s) s1[t] += s1[t + s]; __syncthreads(); }
  const float var = s1[0] * (1.0f / HID);
  const float o = dv * rsqrtf(var + 1e-5f) * g[t] + bt[t];
  if (of) of[(size_t)row * HID + t] = o;
  if (ob) ob[(size_t)row * HID + t] = (__bf16)o;
}

// ---------------------------------------------------------------------------
// Host-side orchestration
// ---------------------------------------------------------------------------
extern "C" void kernel_launch(void* const* d_in, const int* in_sizes, int n_in,
                              void* d_out, int out_size, void* d_ws,
                              size_t ws_size, hipStream_t stream) {
  (void)in_sizes; (void)n_in; (void)out_size; (void)ws_size;

  char* w = (char*)d_ws;
  size_t off = 0;
  auto alloc = [&](size_t bytes) -> void* {
    void* p = (void*)(w + off);
    off = (off + bytes + 255) & ~(size_t)255;
    return p;
  };

  const size_t MB = (size_t)Mrows * HID;   // 9216*256 elems

  // bf16 weight pool (all stored transposed [N,K])
  __bf16* wq    = (__bf16*)alloc(HID * HID * 2);
  __bf16* wk    = (__bf16*)alloc(HID * HID * 2);
  __bf16* wv    = (__bf16*)alloc(HID * HID * 2);
  __bf16* wtg   = (__bf16*)alloc(HID * HID * 2);
  __bf16* wto   = (__bf16*)alloc(HID * HID * 2);
  __bf16* we2q  = (__bf16*)alloc(HID * HID * 2);
  __bf16* wn2eo = (__bf16*)alloc(HID * HID * 2);
  __bf16* wn2k  = (__bf16*)alloc(HID * HID * 2);
  __bf16* wn2v  = (__bf16*)alloc(HID * HID * 2);
  __bf16* wnp   = (__bf16*)alloc((HID + CND) * HID * 2);
  __bf16* wf1   = (__bf16*)alloc(HID * FFN * 2);
  __bf16* wf2   = (__bf16*)alloc(FFN * HID * 2);

  float*  cp    = (float*)alloc(Bsz * HID * 4);
  float*  cg    = (float*)alloc(Bsz * HID * 4);
  __bf16* cfb   = (__bf16*)alloc(MB * 2);
  __bf16* Qb    = (__bf16*)alloc(MB * 2);
  __bf16* Kb    = (__bf16*)alloc(MB * 2);
  __bf16* Vb    = (__bf16*)alloc(MB * 2);
  __bf16* Vt    = (__bf16*)alloc((size_t)Bsz * NH * HD * NNP * 2);
  float*  attf  = (float*)alloc(MB * 4);
  __bf16* attb  = (__bf16*)alloc(MB * 2);
  float*  gatef = (float*)alloc(MB * 4);
  __bf16* attg  = (__bf16*)alloc(MB * 2);
  float*  trif  = (float*)alloc(MB * 4);
  float*  x1f   = (float*)alloc(MB * 4);
  __bf16* x1b   = (__bf16*)alloc(MB * 2);
  float*  qef   = (float*)alloc(MB * 4);
  __bf16* cnb   = (__bf16*)alloc((size_t)Mnode * (HID + CND) * 2);
  __bf16* npjb  = (__bf16*)alloc((size_t)Mnode * HID * 2);
  float*  keysf = (float*)alloc((size_t)Mnode * HID * 4);
  float*  valsf = (float*)alloc((size_t)Mnode * HID * 4);
  float*  aib   = (float*)alloc((size_t)Bsz * Nn * Nn * 4);
  float*  ajb   = (float*)alloc((size_t)Bsz * Nn * Nn * 4);
  __bf16* preb  = (__bf16*)alloc(MB * 2);
  float*  ncf   = (float*)alloc(MB * 4);
  float*  x2f   = (float*)alloc(MB * 4);
  __bf16* x2b   = (__bf16*)alloc(MB * 2);
  __bf16* ffnh  = (__bf16*)alloc((size_t)Mrows * FFN * 2);
  float*  ffno  = (float*)alloc(MB * 4);

  const float* ef   = (const float*)d_in[IN_EF];
  const float* nf   = (const float*)d_in[IN_NF];
  const float* em   = (const float*)d_in[IN_EM];
  const float* nm   = (const float*)d_in[IN_NM];
  const float* cond = (const float*)d_in[IN_COND];
  auto F = [&](int i) { return (const float*)d_in[i]; };

  // 1) weight conversions to transposed bf16 [N,K]
  auto conv = [&](int idx, __bf16* dst, int K, int N) {
    convT_kernel<<<(K * N + 255) / 256, 256, 0, stream>>>(F(idx), dst, K, N);
  };
  conv(P_Q_W, wq, HID, HID);        conv(P_K_W, wk, HID, HID);
  conv(P_V_W, wv, HID, HID);        conv(P_TG_W, wtg, HID, HID);
  conv(P_TO_W, wto, HID, HID);      conv(P_E2Q_W, we2q, HID, HID);
  conv(P_N2EO_W, wn2eo, HID, HID);  conv(P_N2K_W, wn2k, HID, HID);
  conv(P_N2V_W, wn2v, HID, HID);    conv(P_NP_W, wnp, HID + CND, HID);
  conv(P_FFN1_W, wf1, HID, FFN);    conv(P_FFN2_W, wf2, FFN, HID);

  // 2) condition projections + conditioned features (bf16)
  cond_prep_kernel<<<Bsz, HID, 0, stream>>>(cond, F(P_CP_W), F(P_CP_B),
                                            F(P_CG_W), F(P_CG_B), cp, cg);
  build_cf_kernel<<<2048, 256, 0, stream>>>(ef, cp, cg, cfb);

  // 3) Q/K/V GEMMs (bf16 WMMA, 64x64 workgroup tiles)
  dim3 g256(Mrows / 64, HID / 64);
  gemm_bf16_kernel<0><<<g256, 128, 0, stream>>>(cfb, wq, F(P_Q_B), nullptr, Qb, Mrows, HID, HID);
  gemm_bf16_kernel<0><<<g256, 128, 0, stream>>>(cfb, wk, F(P_K_B), nullptr, Kb, Mrows, HID, HID);
  gemm_bf16_kernel<0><<<g256, 128, 0, stream>>>(cfb, wv, F(P_V_B), nullptr, Vb, Mrows, HID, HID);
  repack_vt_kernel<<<2048, 256, 0, stream>>>(Vb, Vt);

  // 4) streaming triangle attention (tri_bias exactly cancels in softmax)
  tri_attn_kernel<<<Bsz * NH * (NN / 16), 32, 0, stream>>>(Qb, Kb, Vt, em, attf, attb);

  // 5) gate + output projection, residual + LN1
  gemm_bf16_kernel<1><<<g256, 128, 0, stream>>>(attb, wtg, F(P_TG_B), gatef, nullptr, Mrows, HID, HID);
  mul_to_bf16_kernel<<<2048, 256, 0, stream>>>(attf, gatef, attg, Mrows * HID);
  gemm_bf16_kernel<0><<<g256, 128, 0, stream>>>(attg, wto, F(P_TO_B), trif, nullptr, Mrows, HID, HID);
  add_ln_kernel<<<Mrows, HID, 0, stream>>>(trif, ef, F(P_LN1G), F(P_LN1B), x1f, x1b);

  // 6) node->edge attention
  gemm_bf16_kernel<0><<<g256, 128, 0, stream>>>(x1b, we2q, F(P_E2Q_B), qef, nullptr, Mrows, HID, HID);
  build_cn_kernel<<<(Mnode * (HID + CND) + 255) / 256, 256, 0, stream>>>(nf, cond, cnb);
  dim3 gn(Mnode / 64, HID / 64);
  gemm_bf16_kernel<0><<<gn, 128, 0, stream>>>(cnb, wnp, F(P_NP_B), nullptr, npjb, Mnode, HID, HID + CND);
  gemm_bf16_kernel<0><<<gn, 128, 0, stream>>>(npjb, wn2k, F(P_N2K_B), keysf, nullptr, Mnode, HID, HID);
  gemm_bf16_kernel<0><<<gn, 128, 0, stream>>>(npjb, wn2v, F(P_N2V_B), valsf, nullptr, Mnode, HID, HID);
  n2e_scores_kernel<<<Bsz * Nn, 64, 0, stream>>>(qef, keysf, nm, aib, ajb);
  build_pre_kernel<<<2048, 256, 0, stream>>>(aib, ajb, valsf, em, preb);
  gemm_bf16_kernel<0><<<g256, 128, 0, stream>>>(preb, wn2eo, F(P_N2EO_B), ncf, nullptr, Mrows, HID, HID);
  add_ln_kernel<<<Mrows, HID, 0, stream>>>(ncf, x1f, F(P_LN2G), F(P_LN2B), x2f, x2b);

  // 7) FFN + LN3 -> output
  dim3 g1024(Mrows / 64, FFN / 64);
  gemm_bf16_kernel<2><<<g1024, 128, 0, stream>>>(x2b, wf1, F(P_FFN1_B), nullptr, ffnh, Mrows, FFN, HID);
  gemm_bf16_kernel<0><<<g256, 128, 0, stream>>>(ffnh, wf2, F(P_FFN2_B), ffno, nullptr, Mrows, HID, FFN);
  add_ln_kernel<<<Mrows, HID, 0, stream>>>(ffno, x2f, F(P_LN3G), F(P_LN3B), (float*)d_out, nullptr);
}
